// TransBlock_77429670412403
// MI455X (gfx1250) — compile-verified, matrix operand order checked by reading, not run on hardware
//
#include <hip/hip_runtime.h>
#include <hip/hip_bf16.h>

// ---------------------------------------------------------------------------
// MI455X (gfx1250): all GEMMs via v_wmma_f32_16x16x32_f16.  Ping-pong
// double-buffered K-loop (unroll 2) -> no register-rotation movs, prefetch
// distance of two K-steps ahead of the matrix pipe.
// ---------------------------------------------------------------------------

typedef _Float16 half_t;
typedef __attribute__((ext_vector_type(16))) _Float16 v16h;
typedef __attribute__((ext_vector_type(8)))  _Float16 v8h;
typedef __attribute__((ext_vector_type(8)))  float    v8f;

#define DIMC   1024
#define HIDDEN 2048
#define NTOK   16384   // B*N
#define SEQ    4096    // N
#define NBH    64      // B*HEADS
#define HD     64      // head dim

// ---- WMMA fragment loader -------------------------------------------------
// 16-bit A/B fragment, wave32: lane L holds row (L&15); within that row the
// 16 halves are K = {k0..k0+7, k0+16..k0+23}, k0 = 8*(L>=16).  Two b128 loads.
__device__ __forceinline__ v16h load_frag(const half_t* __restrict__ p) {
  v8h lo = *(const v8h*)(p);
  v8h hi = *(const v8h*)(p + 16);
  v16h f;
#pragma unroll
  for (int i = 0; i < 8; ++i) { f[i] = lo[i]; f[i + 8] = hi[i]; }
  return f;
}

__device__ __forceinline__ v8f wmma_step(v16h a, v16h b, v8f c) {
  return __builtin_amdgcn_wmma_f32_16x16x32_f16(false, a, false, b,
                                                (short)0, c, false, false);
}

// NT-layout GEMM core: acc[mi][nj] += sum_k A[m,k] * W[n,k]
// One wave -> (16*MT) rows x (16*NT) cols.  K/32 must be even (true for all
// call sites: K in {64, 1024, 2048, 4096}).  Ping-pong buffers, no movs.
template<int MT, int NT>
__device__ __forceinline__ void gemm_core(const half_t* __restrict__ A, int lda,
                                          const half_t* __restrict__ W, int ldw,
                                          int K, int mrow0, int ncol0, int lane,
                                          v8f acc[MT][NT]) {
  const int khi = (lane & 16) ? 8 : 0;
  const half_t* ap[MT];
  const half_t* wp[NT];
#pragma unroll
  for (int i = 0; i < MT; ++i) ap[i] = A + (size_t)(mrow0 + 16 * i) * lda + khi;
#pragma unroll
  for (int j = 0; j < NT; ++j) wp[j] = W + (size_t)(ncol0 + (lane & 15) + 16 * j) * ldw + khi;

  v16h a0[MT], b0[NT], a1[MT], b1[NT];
#pragma unroll
  for (int i = 0; i < MT; ++i) a0[i] = load_frag(ap[i]);
#pragma unroll
  for (int j = 0; j < NT; ++j) b0[j] = load_frag(wp[j]);
#pragma unroll
  for (int i = 0; i < MT; ++i) a1[i] = load_frag(ap[i] + 32);
#pragma unroll
  for (int j = 0; j < NT; ++j) b1[j] = load_frag(wp[j] + 32);

  for (int k = 64; k < K; k += 64) {
#pragma unroll
    for (int i = 0; i < MT; ++i)
#pragma unroll
      for (int j = 0; j < NT; ++j)
        acc[i][j] = wmma_step(a0[i], b0[j], acc[i][j]);
#pragma unroll
    for (int i = 0; i < MT; ++i) a0[i] = load_frag(ap[i] + k);
#pragma unroll
    for (int j = 0; j < NT; ++j) b0[j] = load_frag(wp[j] + k);
#pragma unroll
    for (int i = 0; i < MT; ++i)
#pragma unroll
      for (int j = 0; j < NT; ++j)
        acc[i][j] = wmma_step(a1[i], b1[j], acc[i][j]);
#pragma unroll
    for (int i = 0; i < MT; ++i) a1[i] = load_frag(ap[i] + k + 32);
#pragma unroll
    for (int j = 0; j < NT; ++j) b1[j] = load_frag(wp[j] + k + 32);
  }
#pragma unroll
  for (int i = 0; i < MT; ++i)
#pragma unroll
    for (int j = 0; j < NT; ++j)
      acc[i][j] = wmma_step(a0[i], b0[j], acc[i][j]);
#pragma unroll
  for (int i = 0; i < MT; ++i)
#pragma unroll
    for (int j = 0; j < NT; ++j)
      acc[i][j] = wmma_step(a1[i], b1[j], acc[i][j]);
}

// C/D layout: element r of v8f at lane L -> row = r + 8*(L>=16), col = L&15.

// ---- elementwise helpers ---------------------------------------------------
__global__ __launch_bounds__(256) void cvt_f32_to_f16(const float* __restrict__ in,
                                                      half_t* __restrict__ out, int n) {
  int i = blockIdx.x * 256 + threadIdx.x;
  if (i < n) out[i] = (half_t)in[i];
}

// RMSNorm (torch-faithful): y = x * scale[c] * (sqrt(C)+eps) / ||x||_2
__global__ __launch_bounds__(256) void rmsnorm_f16(const float* __restrict__ x,
                                                   const float* __restrict__ scale,
                                                   half_t* __restrict__ out) {
  __shared__ float red[256];
  const int m = blockIdx.x;
  const float* xr = x + (size_t)m * DIMC;
  float vals[4];
  float ss = 0.f;
#pragma unroll
  for (int i = 0; i < 4; ++i) {
    float v = xr[threadIdx.x + 256 * i];
    vals[i] = v; ss += v * v;
  }
  red[threadIdx.x] = ss; __syncthreads();
  for (int s = 128; s > 0; s >>= 1) {
    if (threadIdx.x < s) red[threadIdx.x] += red[threadIdx.x + s];
    __syncthreads();
  }
  const float f = (32.0f + 1e-6f) / sqrtf(red[0]);   // sqrt(1024)=32
#pragma unroll
  for (int i = 0; i < 4; ++i) {
    int c = threadIdx.x + 256 * i;
    out[(size_t)m * DIMC + c] = (half_t)(vals[i] * scale[c] * f);
  }
}

// ---- QKV GEMM: [16384,1024] x [3072,1024]^T, epilogue phi + head layouts ---
// q  -> [bh][n][d]; kT -> [bh][d][n]; vT -> [bh][d][n]
__global__ __launch_bounds__(256) void qkv_gemm(const half_t* __restrict__ xn,
                                                const half_t* __restrict__ w,
                                                half_t* __restrict__ qb,
                                                half_t* __restrict__ kt,
                                                half_t* __restrict__ vt) {
  const int lane = threadIdx.x & 31, wv = threadIdx.x >> 5;
  const int m_tile = blockIdx.y * 256 + wv * 32;
  const int n_base = blockIdx.x * 64;
  v8f acc[2][4] = {};
  gemm_core<2, 4>(xn, DIMC, w, DIMC, DIMC, m_tile + (lane & 15), n_base, lane, acc);
  const int mhi = (lane & 16) ? 8 : 0;
#pragma unroll
  for (int t = 0; t < 4; ++t) {
    const int f = n_base + 16 * t + (lane & 15);
    const int which = f >> 10, c = f & 1023, hh = c >> 6, dd = c & 63;
#pragma unroll
    for (int mi = 0; mi < 2; ++mi)
#pragma unroll
      for (int r = 0; r < 8; ++r) {
        const int m = m_tile + 16 * mi + r + mhi;
        const int b = m >> 12, n = m & 4095, bh = b * 16 + hh;
        const float v = acc[mi][t][r];
        if (which == 0)
          qb[((size_t)bh * SEQ + n) * HD + dd] = (half_t)__expf(-0.5f * v * v);
        else if (which == 1)
          kt[(size_t)bh * SEQ * HD + (size_t)dd * SEQ + n] = (half_t)__expf(-0.5f * v * v);
        else
          vt[(size_t)bh * SEQ * HD + (size_t)dd * SEQ + n] = (half_t)v;
      }
  }
}

// ---- kv^T GEMM per (b,h): kvT[e,i] = sum_n vT[e,n]*kT[i,n], K=4096 ---------
__global__ __launch_bounds__(256) void kv_gemm(const half_t* __restrict__ kt,
                                               const half_t* __restrict__ vt,
                                               half_t* __restrict__ kvt) {
  const int bh = blockIdx.x;
  const int lane = threadIdx.x & 31, wv = threadIdx.x >> 5;
  const int m_tile = (wv >> 1) * 16;      // e
  const int n_base = (wv & 1) * 32;       // i
  const half_t* A = vt + (size_t)bh * SEQ * HD;
  const half_t* W = kt + (size_t)bh * SEQ * HD;
  v8f acc[1][2] = {};
  gemm_core<1, 2>(A, SEQ, W, SEQ, SEQ, m_tile + (lane & 15), n_base, lane, acc);
  const int mhi = (lane & 16) ? 8 : 0;
#pragma unroll
  for (int t = 0; t < 2; ++t) {
    const int i = n_base + 16 * t + (lane & 15);
#pragma unroll
    for (int r = 0; r < 8; ++r) {
      const int e = m_tile + r + mhi;
      kvt[(size_t)bh * (HD * HD) + e * HD + i] = (half_t)acc[0][t][r];
    }
  }
}

// ---- ksum[bh][i] = sum_n kT[bh][i][n] --------------------------------------
__global__ __launch_bounds__(64) void ksum_kernel(const half_t* __restrict__ kt,
                                                  float* __restrict__ ksum) {
  const int bh = blockIdx.x, i = threadIdx.x;
  const half_t* row = kt + (size_t)bh * SEQ * HD + (size_t)i * SEQ;
  float s = 0.f;
  for (int n = 0; n < SEQ; n += 8) {
    v8h v = *(const v8h*)(row + n);
#pragma unroll
    for (int e = 0; e < 8; ++e) s += (float)v[e];
  }
  ksum[bh * HD + i] = s;
}

// ---- z[bh][n] = 1/(dot(q_n, ksum) + 1e-6) ----------------------------------
__global__ __launch_bounds__(256) void z_kernel(const half_t* __restrict__ qb,
                                                const float* __restrict__ ksum,
                                                float* __restrict__ z) {
  const int bh = blockIdx.y;
  const int n = blockIdx.x * 256 + threadIdx.x;
  const half_t* q = qb + ((size_t)bh * SEQ + n) * HD;
  const float* ks = ksum + bh * HD;
  float s = 0.f;
#pragma unroll 8
  for (int d = 0; d < HD; ++d) s += (float)q[d] * ks[d];
  z[(size_t)bh * SEQ + n] = 1.0f / (s + 1e-6f);
}

// ---- out = q * kvT * z, merged back to [tok, C] f16 ------------------------
__global__ __launch_bounds__(256) void attnout_gemm(const half_t* __restrict__ qb,
                                                    const half_t* __restrict__ kvt,
                                                    const float* __restrict__ z,
                                                    half_t* __restrict__ attnin) {
  const int bh = blockIdx.z;
  const int lane = threadIdx.x & 31, wv = threadIdx.x >> 5;
  const int m_tile = blockIdx.y * 256 + wv * 32;    // token within (b,h)
  const half_t* A = qb  + (size_t)bh * SEQ * HD;
  const half_t* W = kvt + (size_t)bh * (HD * HD);
  v8f acc[2][4] = {};
  gemm_core<2, 4>(A, HD, W, HD, HD, m_tile + (lane & 15), 0, lane, acc);
  const int b = bh >> 4, hh = bh & 15;
  const int mhi = (lane & 16) ? 8 : 0;
#pragma unroll
  for (int t = 0; t < 4; ++t) {
    const int e = 16 * t + (lane & 15);
#pragma unroll
    for (int mi = 0; mi < 2; ++mi)
#pragma unroll
      for (int r = 0; r < 8; ++r) {
        const int n = m_tile + 16 * mi + r + mhi;
        const float v = acc[mi][t][r] * z[(size_t)bh * SEQ + n];
        attnin[((size_t)b * SEQ + n) * DIMC + hh * HD + e] = (half_t)v;
      }
  }
}

// ---- proj GEMM + residual: d_out = x + gamma*(attn@Wp^T + bp) --------------
__global__ __launch_bounds__(256) void proj_gemm(const half_t* __restrict__ attnin,
                                                 const half_t* __restrict__ w,
                                                 const float* __restrict__ bias,
                                                 const float* __restrict__ x,
                                                 const float* __restrict__ gamma,
                                                 float* __restrict__ xout) {
  const int lane = threadIdx.x & 31, wv = threadIdx.x >> 5;
  const int m_tile = blockIdx.y * 256 + wv * 32;
  const int n_base = blockIdx.x * 64;
  v8f acc[2][4] = {};
  gemm_core<2, 4>(attnin, DIMC, w, DIMC, DIMC, m_tile + (lane & 15), n_base, lane, acc);
  const int mhi = (lane & 16) ? 8 : 0;
#pragma unroll
  for (int t = 0; t < 4; ++t) {
    const int o = n_base + 16 * t + (lane & 15);
#pragma unroll
    for (int mi = 0; mi < 2; ++mi)
#pragma unroll
      for (int r = 0; r < 8; ++r) {
        const size_t idx = (size_t)(m_tile + 16 * mi + r + mhi) * DIMC + o;
        xout[idx] = x[idx] + gamma[o] * (acc[mi][t][r] + bias[o]);
      }
  }
}

// ---- FC1 GEMM + exact GELU -------------------------------------------------
__global__ __launch_bounds__(256) void fc1_gemm(const half_t* __restrict__ xn2,
                                                const half_t* __restrict__ w,
                                                const float* __restrict__ bias,
                                                half_t* __restrict__ hbuf) {
  const int lane = threadIdx.x & 31, wv = threadIdx.x >> 5;
  const int m_tile = blockIdx.y * 256 + wv * 32;
  const int n_base = blockIdx.x * 64;
  v8f acc[2][4] = {};
  gemm_core<2, 4>(xn2, DIMC, w, DIMC, DIMC, m_tile + (lane & 15), n_base, lane, acc);
  const int mhi = (lane & 16) ? 8 : 0;
#pragma unroll
  for (int t = 0; t < 4; ++t) {
    const int j = n_base + 16 * t + (lane & 15);
#pragma unroll
    for (int mi = 0; mi < 2; ++mi)
#pragma unroll
      for (int r = 0; r < 8; ++r) {
        const float v = acc[mi][t][r] + bias[j];
        const float g = 0.5f * v * (1.0f + erff(v * 0.70710678118654752f));
        hbuf[(size_t)(m_tile + 16 * mi + r + mhi) * HIDDEN + j] = (half_t)g;
      }
  }
}

// ---- FC2 GEMM + residual into d_out (holds x2) -----------------------------
__global__ __launch_bounds__(256) void fc2_gemm(const half_t* __restrict__ hbuf,
                                                const half_t* __restrict__ w,
                                                const float* __restrict__ bias,
                                                const float* __restrict__ gamma,
                                                float* __restrict__ out) {
  const int lane = threadIdx.x & 31, wv = threadIdx.x >> 5;
  const int m_tile = blockIdx.y * 256 + wv * 32;
  const int n_base = blockIdx.x * 64;
  v8f acc[2][4] = {};
  gemm_core<2, 4>(hbuf, HIDDEN, w, HIDDEN, HIDDEN, m_tile + (lane & 15), n_base, lane, acc);
  const int mhi = (lane & 16) ? 8 : 0;
#pragma unroll
  for (int t = 0; t < 4; ++t) {
    const int c = n_base + 16 * t + (lane & 15);
#pragma unroll
    for (int mi = 0; mi < 2; ++mi)
#pragma unroll
      for (int r = 0; r < 8; ++r) {
        const size_t idx = (size_t)(m_tile + 16 * mi + r + mhi) * DIMC + c;
        out[idx] = out[idx] + gamma[c] * (acc[mi][t][r] + bias[c]);
      }
  }
}

// ---------------------------------------------------------------------------
extern "C" void kernel_launch(void* const* d_in, const int* in_sizes, int n_in,
                              void* d_out, int out_size, void* d_ws, size_t ws_size,
                              hipStream_t stream) {
  const float* x      = (const float*)d_in[0];
  const float* scale1 = (const float*)d_in[1];
  const float* scale2 = (const float*)d_in[2];
  const float* gamma  = (const float*)d_in[3];
  const float* qkv_w  = (const float*)d_in[4];
  const float* proj_w = (const float*)d_in[5];
  const float* proj_b = (const float*)d_in[6];
  const float* c1_w   = (const float*)d_in[7];
  const float* c1_b   = (const float*)d_in[8];
  const float* c2_w   = (const float*)d_in[9];
  const float* c2_b   = (const float*)d_in[10];
  float* out = (float*)d_out;

  // workspace layout (256B aligned, sizes are multiples of 256)
  char* ws = (char*)d_ws;
  size_t off = 0;
  auto take = [&](size_t bytes) { char* p = ws + off; off += (bytes + 255) & ~(size_t)255; return p; };
  half_t* wq  = (half_t*)take((size_t)3 * DIMC * DIMC * 2);      // qkv_w f16
  half_t* wp  = (half_t*)take((size_t)DIMC * DIMC * 2);          // proj_w f16
  half_t* w1  = (half_t*)take((size_t)HIDDEN * DIMC * 2);        // conv1_w f16
  half_t* w2  = (half_t*)take((size_t)DIMC * HIDDEN * 2);        // conv2_w f16
  half_t* xn  = (half_t*)take((size_t)NTOK * DIMC * 2);          // xn / xn2
  half_t* qb  = (half_t*)take((size_t)NBH * SEQ * HD * 2);       // phi(q)
  half_t* kt  = (half_t*)take((size_t)NBH * SEQ * HD * 2);       // phi(k)^T
  half_t* vt  = (half_t*)take((size_t)NBH * SEQ * HD * 2);       // v^T / attnin
  half_t* kvt = (half_t*)take((size_t)NBH * HD * HD * 2);
  float*  ksm = (float*)take((size_t)NBH * HD * 4);
  float*  zb  = (float*)take((size_t)NBH * SEQ * 4);
  half_t* attnin = vt;          // reuse: vT dead after kv_gemm
  half_t* hbuf   = qb;          // reuse: qb+kt (64MB contiguous) dead after attnout

  // 1) weights -> f16
  cvt_f32_to_f16<<<(3 * DIMC * DIMC + 255) / 256, 256, 0, stream>>>(qkv_w, wq, 3 * DIMC * DIMC);
  cvt_f32_to_f16<<<(DIMC * DIMC + 255) / 256, 256, 0, stream>>>(proj_w, wp, DIMC * DIMC);
  cvt_f32_to_f16<<<(HIDDEN * DIMC + 255) / 256, 256, 0, stream>>>(c1_w, w1, HIDDEN * DIMC);
  cvt_f32_to_f16<<<(DIMC * HIDDEN + 255) / 256, 256, 0, stream>>>(c2_w, w2, DIMC * HIDDEN);

  // 2) rmsnorm1
  rmsnorm_f16<<<NTOK, 256, 0, stream>>>(x, scale1, xn);
  // 3) qkv + phi + head layouts
  qkv_gemm<<<dim3(3 * DIMC / 64, NTOK / 256), 256, 0, stream>>>(xn, wq, qb, kt, vt);
  // 4) kv^T per (b,h)
  kv_gemm<<<NBH, 256, 0, stream>>>(kt, vt, kvt);
  // 5) k row-sums
  ksum_kernel<<<NBH, 64, 0, stream>>>(kt, ksm);
  // 6) z
  z_kernel<<<dim3(SEQ / 256, NBH), 256, 0, stream>>>(qb, ksm, zb);
  // 7) out = q * kvT * z  -> attnin [tok, C]
  attnout_gemm<<<dim3(1, SEQ / 256, NBH), 256, 0, stream>>>(qb, kvt, zb, attnin);
  // 8) proj + residual -> d_out holds x2
  proj_gemm<<<dim3(DIMC / 64, NTOK / 256), 256, 0, stream>>>(attnin, wp, proj_b, x, gamma, out);
  // 9) rmsnorm2 (reads d_out)
  rmsnorm_f16<<<NTOK, 256, 0, stream>>>(out, scale2, xn);
  // 10) fc1 + gelu
  fc1_gemm<<<dim3(HIDDEN / 64, NTOK / 256), 256, 0, stream>>>(xn, w1, c1_b, hbuf);
  // 11) fc2 + residual into d_out
  fc2_gemm<<<dim3(DIMC / 64, NTOK / 256), 256, 0, stream>>>(hbuf, w2, c2_b, gamma, out);

  (void)in_sizes; (void)n_in; (void)out_size; (void)ws_size;
}
